// AttentiveTransformer_65197603554279
// MI455X (gfx1250) — matrix operand order repaired and checked by reading, not compiled, not run on hardware
//
#include <hip/hip_runtime.h>

// Problem constants (from reference setup_inputs)
#define NB      32768      // batch rows
#define DIN     2048       // input features (K)
#define DOUT    1024       // output features (N)
#define VBS     128        // ghost-batch virtual batch size == our M tile
#define KSTEP   32         // bf16 WMMA K per step
#define KITERS  (DIN / KSTEP)   // 64

typedef __bf16        v16bf __attribute__((ext_vector_type(16)));
typedef float         v8f   __attribute__((ext_vector_type(8)));
typedef unsigned int  v8u   __attribute__((ext_vector_type(8)));

__device__ __forceinline__ unsigned short f32_to_bf16(float f) {
    unsigned int u = __float_as_uint(f);
    u += 0x7FFFu + ((u >> 16) & 1u);          // round-to-nearest-even
    return (unsigned short)(u >> 16);
}

// Build a 16x32 bf16 WMMA fragment from LDS: two 16B chunks per lane.
__device__ __forceinline__ v16bf frag_from_lds(const unsigned short* s, int o0, int o1) {
    uint4 a = *(const uint4*)(s + o0);
    uint4 b = *(const uint4*)(s + o1);
    v8u u = { a.x, a.y, a.z, a.w, b.x, b.y, b.z, b.w };
    return __builtin_bit_cast(v16bf, u);
}

// ---------------------------------------------------------------------------
// Kernel 1: fused  x = feat @ W^T  ->  GhostBN(128-row virtual batches)
//                  -> x *= priors  -> write to out (pre-sparsemax values)
// Grid: (DOUT/128, NB/128); Block: 256 threads (8 wave32 in a 2x4 grid).
// Each workgroup owns exactly ONE virtual batch x 128 columns, so the GBN
// statistics live entirely in registers + a tiny LDS array in the epilogue.
//
// Main loop is software-pipelined through registers: global loads for
// K-tile (k+1) are issued BEFORE the 8-WMMA burst for K-tile k; the
// s_wait_loadcnt lands at the next iteration's convert/ds_store, so VMEM
// latency overlaps the entire WMMA + fragment-load phase.
// ---------------------------------------------------------------------------
__global__ __launch_bounds__(256)
void fused_gemm_gbn_kernel(const float* __restrict__ A,      // [NB, DIN]
                           const float* __restrict__ W,      // [DOUT, DIN]
                           const float* __restrict__ gamma,  // [DOUT]
                           const float* __restrict__ beta,   // [DOUT]
                           const float* __restrict__ priors, // [NB, DOUT]
                           float* __restrict__ out)          // [NB, DOUT]
{
    __shared__ unsigned short sA[VBS * KSTEP];   // 128x32 bf16  (8 KB)
    __shared__ unsigned short sB[128 * KSTEP];   // 128x32 bf16  (8 KB)
    __shared__ float s_sum[2][128];
    __shared__ float s_sq [2][128];

    const int tid   = threadIdx.x;
    const int lane  = tid & 31;
    const int wave  = tid >> 5;
    const int wr    = wave >> 2;      // 0..1 : wave row  (64 rows each)
    const int wc    = wave & 3;       // 0..3 : wave col  (32 cols each)
    const int l15   = lane & 15;
    const int khalf = lane >> 4;      // 0/1

    const int rowbase = blockIdx.y * VBS;    // virtual-batch start row
    const int colbase = blockIdx.x * 128;    // output-column tile start

    // Per-thread fixed staging coordinates: chunk i covers (row r_i, 4 cols c4_i).
    int rr[4], cc[4];
    const float* pA[4];
    const float* pW[4];
    #pragma unroll
    for (int i = 0; i < 4; ++i) {
        const int chunk = tid + 256 * i;       // 0..1023
        rr[i] = chunk >> 3;                    // 0..127
        cc[i] = (chunk & 7) << 2;              // 0,4,..,28
        pA[i] = A + (size_t)(rowbase + rr[i]) * DIN + cc[i];
        pW[i] = W + (size_t)(colbase + rr[i]) * DIN + cc[i];
    }

    v8f acc[4][2];
    #pragma unroll
    for (int mt = 0; mt < 4; ++mt)
        #pragma unroll
        for (int nt = 0; nt < 2; ++nt)
            acc[mt][nt] = (v8f){0.f,0.f,0.f,0.f,0.f,0.f,0.f,0.f};

    // Prologue: fetch K-tile 0 into registers.
    float4 avr[4], wvr[4];
    #pragma unroll
    for (int i = 0; i < 4; ++i) {
        avr[i] = *(const float4*)(pA[i]);
        wvr[i] = *(const float4*)(pW[i]);
    }

    for (int kk = 0; kk < KITERS; ++kk) {
        __syncthreads();   // previous iteration's fragment reads are done

        // Convert + store the registered K-tile kk into LDS.
        #pragma unroll
        for (int i = 0; i < 4; ++i) {
            uint2 ap, wp;
            ap.x = (unsigned)f32_to_bf16(avr[i].x) | ((unsigned)f32_to_bf16(avr[i].y) << 16);
            ap.y = (unsigned)f32_to_bf16(avr[i].z) | ((unsigned)f32_to_bf16(avr[i].w) << 16);
            wp.x = (unsigned)f32_to_bf16(wvr[i].x) | ((unsigned)f32_to_bf16(wvr[i].y) << 16);
            wp.y = (unsigned)f32_to_bf16(wvr[i].z) | ((unsigned)f32_to_bf16(wvr[i].w) << 16);
            *(uint2*)(sA + rr[i] * KSTEP + cc[i]) = ap;
            *(uint2*)(sB + rr[i] * KSTEP + cc[i]) = wp;
        }
        __syncthreads();

        // Issue global loads for K-tile kk+1 now; they complete behind the
        // WMMA burst below (consumed only at the next iteration's stores).
        if (kk + 1 < KITERS) {
            const int ko = (kk + 1) * KSTEP;
            #pragma unroll
            for (int i = 0; i < 4; ++i) {
                avr[i] = *(const float4*)(pA[i] + ko);
                wvr[i] = *(const float4*)(pW[i] + ko);
            }
        }

        // Fragments per ISA 16-bit layouts (A: chunks at khalf*8 and 16+khalf*8;
        // B: lane = column, same K chunks as A: khalf*8 and 16+khalf*8).
        v16bf aF[4], bF[2];
        #pragma unroll
        for (int mt = 0; mt < 4; ++mt) {
            const int r = wr * 64 + mt * 16 + l15;
            aF[mt] = frag_from_lds(sA, r * KSTEP + khalf * 8, r * KSTEP + 16 + khalf * 8);
        }
        #pragma unroll
        for (int nt = 0; nt < 2; ++nt) {
            const int c = wc * 32 + nt * 16 + l15;
            bF[nt] = frag_from_lds(sB, c * KSTEP + khalf * 8, c * KSTEP + 16 + khalf * 8);
        }
        #pragma unroll
        for (int mt = 0; mt < 4; ++mt)
            #pragma unroll
            for (int nt = 0; nt < 2; ++nt)
                acc[mt][nt] = __builtin_amdgcn_wmma_f32_16x16x32_bf16(
                    false, aF[mt], false, bF[nt], (short)0, acc[mt][nt], false, false);
    }

    // ---- Epilogue: Ghost BatchNorm stats over the 128 rows of this tile ----
    // Per-wave column sums across its 64 rows (8 regs + shfl_xor(16)).
    #pragma unroll
    for (int nt = 0; nt < 2; ++nt) {
        float cs = 0.f, cq = 0.f;
        #pragma unroll
        for (int mt = 0; mt < 4; ++mt) {
            v8f a = acc[mt][nt];
            #pragma unroll
            for (int r = 0; r < 8; ++r) { cs += a[r]; cq += a[r] * a[r]; }
        }
        cs += __shfl_xor(cs, 16);
        cq += __shfl_xor(cq, 16);
        if (lane < 16) {
            s_sum[wr][wc * 32 + nt * 16 + lane] = cs;
            s_sq [wr][wc * 32 + nt * 16 + lane] = cq;
        }
    }
    __syncthreads();

    #pragma unroll
    for (int nt = 0; nt < 2; ++nt) {
        const int cl   = wc * 32 + nt * 16 + l15;
        const int gcol = colbase + cl;
        const float mean = (s_sum[0][cl] + s_sum[1][cl]) * (1.f / 128.f);
        const float ex2  = (s_sq [0][cl] + s_sq [1][cl]) * (1.f / 128.f);
        const float rstd = rsqrtf(ex2 - mean * mean + 1e-5f);
        const float g    = gamma[gcol] * rstd;
        const float b    = beta[gcol] - mean * g;
        #pragma unroll
        for (int mt = 0; mt < 4; ++mt) {
            v8f a = acc[mt][nt];
            #pragma unroll
            for (int r = 0; r < 8; ++r) {
                const int row = rowbase + wr * 64 + mt * 16 + r + ((lane >= 16) ? 8 : 0);
                const float v = (a[r] * g + b) * priors[(size_t)row * DOUT + gcol];
                out[(size_t)row * DOUT + gcol] = v;
            }
        }
    }
}

// ---------------------------------------------------------------------------
// Kernel 2: sparsemax over each row of out[NB, DOUT], in place.
// tau solves sum(max(z - tau, 0)) == 1 (monotone) -> bisection on an
// interval of initial width 1.0; 32 steps reach fp32 resolution, no sort.
// ---------------------------------------------------------------------------
__global__ __launch_bounds__(256)
void sparsemax_kernel(float* __restrict__ x)
{
    __shared__ float sred[8];
    __shared__ float sbc;

    float* p = x + (size_t)blockIdx.x * DOUT;
    const int tid  = threadIdx.x;
    const int lane = tid & 31;
    const int wv   = tid >> 5;

    float4 z = ((const float4*)p)[tid];   // 4 elements per thread

    // row max
    float m = fmaxf(fmaxf(z.x, z.y), fmaxf(z.z, z.w));
    #pragma unroll
    for (int off = 16; off >= 1; off >>= 1) m = fmaxf(m, __shfl_xor(m, off));
    if (lane == 0) sred[wv] = m;
    __syncthreads();
    if (tid == 0) {
        float mm = sred[0];
        #pragma unroll
        for (int i = 1; i < 8; ++i) mm = fmaxf(mm, sred[i]);
        sbc = mm;
    }
    __syncthreads();
    const float rowmax = sbc;

    float lo = rowmax - 1.0f, hi = rowmax;   // f(lo) >= 1 >= f(hi)
    for (int it = 0; it < 32; ++it) {
        const float tau = 0.5f * (lo + hi);
        float s = fmaxf(z.x - tau, 0.f) + fmaxf(z.y - tau, 0.f)
                + fmaxf(z.z - tau, 0.f) + fmaxf(z.w - tau, 0.f);
        #pragma unroll
        for (int off = 16; off >= 1; off >>= 1) s += __shfl_xor(s, off);
        if (lane == 0) sred[wv] = s;
        __syncthreads();
        if (tid == 0) {
            float t = sred[0];
            #pragma unroll
            for (int i = 1; i < 8; ++i) t += sred[i];
            sbc = t;
        }
        __syncthreads();
        const float tot = sbc;
        if (tot >= 1.0f) lo = tau; else hi = tau;
    }
    const float tau = 0.5f * (lo + hi);

    z.x = fmaxf(z.x - tau, 0.f);
    z.y = fmaxf(z.y - tau, 0.f);
    z.z = fmaxf(z.z - tau, 0.f);
    z.w = fmaxf(z.w - tau, 0.f);
    ((float4*)p)[tid] = z;
}

extern "C" void kernel_launch(void* const* d_in, const int* in_sizes, int n_in,
                              void* d_out, int out_size, void* d_ws, size_t ws_size,
                              hipStream_t stream) {
    (void)in_sizes; (void)n_in; (void)out_size; (void)d_ws; (void)ws_size;
    const float* priors = (const float*)d_in[0];
    const float* feat   = (const float*)d_in[1];
    const float* W      = (const float*)d_in[2];
    const float* gamma  = (const float*)d_in[3];
    const float* beta   = (const float*)d_in[4];
    float* out = (float*)d_out;

    dim3 g1(DOUT / 128, NB / VBS);   // (8, 256)
    fused_gemm_gbn_kernel<<<g1, 256, 0, stream>>>(feat, W, gamma, beta, priors, out);
    sparsemax_kernel<<<NB, 256, 0, stream>>>(out);
}
